// SIGEAutoencoder3d_80951543595461
// MI455X (gfx1250) — compile-verified, matrix operand order checked by reading, not run on hardware
//
#include <hip/hip_runtime.h>
#include <math.h>

// ---------------------------------------------------------------------------
// CDNA5 WMMA (wave32) types: bf16 A/B fragments (16 bf16 = 8 VGPRs),
// f32 C/D accumulator (8 VGPRs).
// ---------------------------------------------------------------------------
typedef __attribute__((ext_vector_type(16))) __bf16         bf16x16;
typedef __attribute__((ext_vector_type(8)))  float          v8f;
typedef __attribute__((ext_vector_type(8)))  unsigned short us8;

__device__ __forceinline__ unsigned short f2bf(float f) {
  // round-to-nearest-even fp32 -> bf16 (bit-level)
  unsigned int u = __float_as_uint(f);
  u += 0x7FFFu + ((u >> 16) & 1u);
  return (unsigned short)(u >> 16);
}

// ---------------------------------------------------------------------------
// Weight repack: OIDHW fp32 -> per-lane WMMA B-fragment stream (bf16).
// K order = tap-major (kt,ky,kx), 32 input channels per chunk (matches the
// conv kernel's K loop).  Fragment mapping (ISA 7.12.2, 16-bit 32x16 B):
//   lane -> column n = nt*16 + (lane&15); half = lane>>4
//   element e in 0..7  -> k = half*8 + e
//   element e in 8..15 -> k = 16 + half*8 + (e-8)
// Output: wpk[((nt*KC + c)*32 + lane)*16 + e], zero-padded for cout/cin tails.
// ---------------------------------------------------------------------------
__global__ void repack_w(const float* __restrict__ w, unsigned short* __restrict__ wpk,
                         int cin, int cout, int k3, int NT, int KC)
{
  int t = blockIdx.x * blockDim.x + threadIdx.x;
  int total = NT * KC * 32;
  if (t >= total) return;
  int lane = t & 31;
  int c    = (t >> 5) % KC;
  int nt   = t / (32 * KC);
  int cpc  = KC / k3;
  int tap  = c / cpc;
  int ci0  = (c - tap * cpc) << 5;
  int l16  = lane & 15, half = lane >> 4;
  int co   = nt * 16 + l16;

  unsigned short vals[16];
  #pragma unroll
  for (int e = 0; e < 16; ++e) {
    int k  = (e < 8) ? (half * 8 + e) : (16 + half * 8 + (e - 8));
    int ci = ci0 + k;
    float v = 0.f;
    if (co < cout && ci < cin)
      v = w[(size_t)(co * cin + ci) * k3 + tap];
    vals[e] = f2bf(v);
  }
  us8* dst = (us8*)(wpk + (size_t)t * 16);
  dst[0] = *(const us8*)&vals[0];
  dst[1] = *(const us8*)&vals[8];
}

// ---------------------------------------------------------------------------
// Implicit-GEMM 3D causal conv using v_wmma_f32_16x16x32_bf16.
// Block tile: M = 64 sites x N = 64*NTW output channels (4 waves).
// Each wave: NTW n-tiles x 4 m-tiles -> 4*NTW WMMAs per 32-wide K chunk.
// B fragments register-resident (2 x global_load_b128 each, repacked stream);
// A staged through LDS with a branch-free, compare-free masked gather
// (addresses clamped per tap, zeros realized by multiplying a {0,1} mask).
// ---------------------------------------------------------------------------
template <int NTW>
__global__ __launch_bounds__(128) void conv3d_wmma(
    const float* __restrict__ in, const unsigned short* __restrict__ wpk,
    const float* __restrict__ bias, const float* __restrict__ resid,
    float* __restrict__ out, int cin, int cout, int T, int H, int W, int ks)
{
  // rows padded to 40 ushorts (80 B, 16B-aligned) -> b128 fragment loads
  __shared__ __align__(16) unsigned short As[64][40];   // [m][k]
  __shared__ int st[64], sy[64], sx[64];                // site coords (once)
  __shared__ int boffs[64];                             // per-tap offset, -1 = pad

  const int S    = T * H * W;
  const int HW   = H * W;
  const int s0   = blockIdx.x << 6;     // 64 sites per block (S % 64 == 0)
  const int tid  = threadIdx.x;
  const int wave = tid >> 5;
  const int lane = tid & 31;
  const int half = lane >> 4;
  const int l16  = lane & 15;
  const int kk   = tid & 31;            // this thread's fixed ci sub-index
  const int pad  = ks >> 1;             // spatial SAME
  const int tpad = ks - 1;              // causal: past frames only
  const int cpc  = (cin + 31) >> 5;
  const int KC   = ks * ks * ks * cpc;

  if (tid < 64) {                       // decode 64 site coords once
    int site = s0 + tid;
    int t = site / HW; int r = site - t * HW;
    int y = r / W;
    st[tid] = t; sy[tid] = y; sx[tid] = r - y * W;
  }

  v8f acc[NTW][4];
  #pragma unroll
  for (int i = 0; i < NTW; ++i)
    #pragma unroll
    for (int mi = 0; mi < 4; ++mi)
      acc[i][mi] = (v8f){0.f,0.f,0.f,0.f,0.f,0.f,0.f,0.f};

  // this wave's global 16-wide n-tiles (spaced by 4 so waves stay contiguous)
  const unsigned short* bbase[NTW];
  #pragma unroll
  for (int i = 0; i < NTW; ++i) {
    int ntg = blockIdx.y * 4 * NTW + 4 * i + wave;
    bbase[i] = wpk + (size_t)ntg * KC * 512;
  }

  int c = 0;
  for (int kt = 0; kt < ks; ++kt)
  for (int ky = 0; ky < ks; ++ky)
  for (int kx = 0; kx < ks; ++kx) {
    __syncthreads();                    // prior As/boffs consumers done
    if (tid < 64) {
      int ti = st[tid] + kt - tpad;     // ti <= t < T always (causal)
      int yi = sy[tid] + ky - pad;
      int xi = sx[tid] + kx - pad;
      boffs[tid] = (ti >= 0 && yi >= 0 && yi < H && xi >= 0 && xi < W)
                       ? (ti * HW + yi * W + xi) : -1;
    }
    __syncthreads();
    // clamped offsets (always legal addresses) + {0,1} float masks
    int   off[16];
    float fm[16];
    #pragma unroll
    for (int j = 0; j < 16; ++j) {
      int b  = boffs[(j << 2) + wave];
      off[j] = (b >= 0) ? b : 0;
      fm[j]  = (b >= 0) ? 1.f : 0.f;
    }

    for (int ci0 = 0; ci0 < cin; ci0 += 32, ++c) {
      __syncthreads();                  // prior fragment reads of As done
      // ---- stage A: branch-free masked gather, 16 elements/thread ----
      const int   ci   = ci0 + kk;
      const float cmul = (ci < cin) ? 1.f : 0.f;   // only conv_in tail
      const float* ip  = in + (size_t)(ci < cin ? ci : 0) * S;
      #pragma unroll
      for (int j = 0; j < 16; ++j) {
        float v = ip[off[j]] * (fm[j] * cmul);
        As[(j << 2) + wave][kk] = f2bf(v);
      }
      __syncthreads();

      // ---- B fragments: two aligned b128 loads each from repacked stream ----
      union { bf16x16 v; us8 h[2]; } B[NTW];
      #pragma unroll
      for (int i = 0; i < NTW; ++i) {
        const us8* bp = (const us8*)(bbase[i] + ((size_t)c * 32 + lane) * 16);
        B[i].h[0] = bp[0];
        B[i].h[1] = bp[1];
      }

      // ---- 4 m-tiles x NTW n-tiles of WMMA; A fragment reused over n ----
      #pragma unroll
      for (int mi = 0; mi < 4; ++mi) {
        union { bf16x16 v; us8 h[2]; } A;
        A.h[0] = *(const us8*)&As[(mi << 4) + l16][half << 3];
        A.h[1] = *(const us8*)&As[(mi << 4) + l16][16 + (half << 3)];
        #pragma unroll
        for (int i = 0; i < NTW; ++i)
          acc[i][mi] = __builtin_amdgcn_wmma_f32_16x16x32_bf16(
              false, A.v, false, B[i].v, (short)0, acc[i][mi], false, false);
      }
    }
  }

  // ---- epilogue: bias + optional residual ----
  // C/D layout: VGPR r -> M = r + half*8, N = l16.
  #pragma unroll
  for (int i = 0; i < NTW; ++i) {
    const int co = (blockIdx.y * 4 * NTW + 4 * i + wave) * 16 + l16;
    if (co < cout) {
      const float bv = bias[co];
      #pragma unroll
      for (int mi = 0; mi < 4; ++mi) {
        #pragma unroll
        for (int r = 0; r < 8; ++r) {
          int site = s0 + (mi << 4) + r + (half << 3);
          float v  = acc[i][mi][r] + bv;
          if (resid) v += resid[(size_t)co * S + site];
          out[(size_t)co * S + site] = v;
        }
      }
    }
  }
}

// ---------------------------------------------------------------------------
// Channel-only RMSNorm (F.normalize over C * sqrt(C) * gamma) fused with SiLU.
// ---------------------------------------------------------------------------
__global__ void rmsnorm_silu(const float* __restrict__ in,
                             const float* __restrict__ g,
                             float* __restrict__ out, int C, int S)
{
  int s = blockIdx.x * blockDim.x + threadIdx.x;
  if (s >= S) return;
  float ss = 0.f;
  for (int c = 0; c < C; ++c) {
    float v = in[(size_t)c * S + s];
    ss += v * v;
  }
  float scale = sqrtf((float)C) / fmaxf(sqrtf(ss), 1e-12f);
  for (int c = 0; c < C; ++c) {
    float v = in[(size_t)c * S + s] * scale * g[c];
    out[(size_t)c * S + s] = v / (1.f + __expf(-v));   // silu
  }
}

// ---------------------------------------------------------------------------
// Nearest-neighbor (1,2,2) upsample.
// ---------------------------------------------------------------------------
__global__ void upsample2x(const float* __restrict__ in, float* __restrict__ out,
                           int C, int T, int H, int W)
{
  size_t total = (size_t)C * T * (2 * H) * (2 * W);
  size_t idx = (size_t)blockIdx.x * blockDim.x + threadIdx.x;
  if (idx >= total) return;
  int W2 = 2 * W, H2 = 2 * H;
  int x = (int)(idx % W2); size_t r = idx / W2;
  int y = (int)(r % H2);   r /= H2;
  int t = (int)(r % T);
  int c = (int)(r / T);
  out[idx] = in[(((size_t)c * T + t) * H + (y >> 1)) * W + (x >> 1)];
}

// ---------------------------------------------------------------------------
// Host orchestration. Input order = jax.tree_leaves of setup_inputs()
// (dicts flattened with sorted keys, lists in order); if the harness puts z
// first (insertion order), in_sizes[0]==16384 and the param base shifts by 1.
// ---------------------------------------------------------------------------
extern "C" void kernel_launch(void* const* d_in, const int* in_sizes, int n_in,
                              void* d_out, int out_size, void* d_ws, size_t ws_size,
                              hipStream_t stream)
{
  (void)out_size; (void)ws_size;
  const int z_first = (n_in > 0 && in_sizes[0] == 4 * 4 * 32 * 32);
  const int pb = z_first ? 1 : 0;
  const float* Z = (const float*)d_in[z_first ? 0 : 61];
  auto P = [&](int i) { return (const float*)d_in[pb + i]; };

  // 4 fp32 ping-pong buffers (each 128ch x 4 x 128 x 128) + packed-weight region
  const size_t BUF = (size_t)128 * 4 * 128 * 128;        // 8,388,608 floats
  float* b0 = (float*)d_ws;
  float* b1 = b0 + BUF;
  float* b2 = b1 + BUF;                                  // norm scratch
  float* b3 = b2 + BUF;                                  // conv1 scratch
  unsigned short* wpk = (unsigned short*)(b3 + BUF);     // <= 3.54 MB used

  auto conv = [&](const float* in, const float* wt, const float* bi,
                  const float* resid, float* out,
                  int cin, int cout, int T, int H, int W, int ks) {
    int S   = T * H * W;
    int k3  = ks * ks * ks;
    int cpc = (cin + 31) / 32;
    int KC  = k3 * cpc;
    int ntw = (cout >= 128) ? 2 : 1;                     // n-tiles per wave
    int ngb = (cout + 64 * ntw - 1) / (64 * ntw);        // grid.y
    int NT  = ngb * 4 * ntw;                             // padded 16-wide tiles
    int total = NT * KC * 32;
    repack_w<<<(total + 255) / 256, 256, 0, stream>>>(wt, wpk, cin, cout, k3, NT, KC);
    dim3 grid(S / 64, ngb);
    if (ntw == 2)
      conv3d_wmma<2><<<grid, dim3(128), 0, stream>>>(in, wpk, bi, resid, out,
                                                     cin, cout, T, H, W, ks);
    else
      conv3d_wmma<1><<<grid, dim3(128), 0, stream>>>(in, wpk, bi, resid, out,
                                                     cin, cout, T, H, W, ks);
  };
  auto norm = [&](const float* in, const float* g, float* out, int C, int S) {
    rmsnorm_silu<<<(S + 255) / 256, 256, 0, stream>>>(in, g, out, C, S);
  };
  auto upsmp = [&](const float* in, float* out, int C, int T, int H, int W) {
    size_t total = (size_t)C * T * 4 * H * W;
    upsample2x<<<(unsigned)((total + 255) / 256), 256, 0, stream>>>(in, out, C, T, H, W);
  };
  auto resblock = [&](const float* x, float* outbuf, int cin, int cout,
                      int T, int H, int W,
                      int c1b, int c1w, int c2b, int c2w, int g1, int g2,
                      int ninb, int ninw) {
    int S = T * H * W;
    norm(x, P(g1), b2, cin, S);
    conv(b2, P(c1w), P(c1b), nullptr, b3, cin, cout, T, H, W, 3);
    norm(b3, P(g2), b2, cout, S);
    const float* resid;
    if (ninw >= 0) {   // 1x1x1 nin shortcut, also via WMMA
      conv(x, P(ninw), P(ninb), nullptr, outbuf, cin, cout, T, H, W, 1);
      resid = outbuf;  // each lane reads then writes its own element: safe
    } else {
      resid = x;
    }
    conv(b2, P(c2w), P(c2b), resid, outbuf, cout, cout, T, H, W, 3);
  };

  int T = 4, H = 32, W = 32;

  // conv_in: 4 -> 256 (K zero-padded to 32)
  conv(Z, P(1), P(0), nullptr, b0, 4, 256, T, H, W, 3);

  // mid1, mid2 @256
  resblock(b0, b1, 256, 256, T, H, W, 4, 5, 6, 7, 8, 9, -1, -1);
  resblock(b1, b0, 256, 256, T, H, W, 10, 11, 12, 13, 14, 15, -1, -1);

  // up level 2: two 256->256 blocks, upsample, up_conv 256->256
  resblock(b0, b1, 256, 256, T, H, W, 47, 48, 49, 50, 51, 52, -1, -1);
  resblock(b1, b0, 256, 256, T, H, W, 53, 54, 55, 56, 57, 58, -1, -1);
  upsmp(b0, b1, 256, T, H, W); H *= 2; W *= 2;           // 64x64
  conv(b1, P(60), P(59), nullptr, b0, 256, 256, T, H, W, 3);

  // up level 1: 256->128 (nin), 128->128, upsample, up_conv 128->128
  resblock(b0, b1, 256, 128, T, H, W, 31, 32, 33, 34, 35, 36, 37, 38);
  resblock(b1, b0, 128, 128, T, H, W, 39, 40, 41, 42, 43, 44, -1, -1);
  upsmp(b0, b1, 128, T, H, W); H *= 2; W *= 2;           // 128x128
  conv(b1, P(46), P(45), nullptr, b0, 128, 128, T, H, W, 3);

  // up level 0: 128->64 (nin), 64->64
  resblock(b0, b1, 128, 64, T, H, W, 17, 18, 19, 20, 21, 22, 23, 24);
  resblock(b1, b0, 64, 64, T, H, W, 25, 26, 27, 28, 29, 30, -1, -1);

  // norm_out + silu, conv_out 64 -> 3 (cout tail predicated in kernel)
  int S = T * H * W;
  norm(b0, P(16), b2, 64, S);
  conv(b2, P(3), P(2), nullptr, (float*)d_out, 64, 3, T, H, W, 3);
}
